// MultiHeadAttention_90477781058198
// MI455X (gfx1250) — matrix operand order repaired and checked by reading, not compiled, not run on hardware
//
#include <hip/hip_runtime.h>

typedef __attribute__((ext_vector_type(16))) _Float16 v16h;
typedef __attribute__((ext_vector_type(8)))  _Float16 v8h;
typedef __attribute__((ext_vector_type(4)))  _Float16 v4h;
typedef __attribute__((ext_vector_type(8)))  float    v8f;

#define D_MODEL   1024
#define NUM_HEADS 16
#define D_K       64
#define BATCH     2
#define SEQ       2048
#define M_ROWS    (BATCH * SEQ)   // 4096

// ---------------------------------------------------------------------------
// WMMA helpers (CDNA5 16x16x32 f16 -> f32)
// ---------------------------------------------------------------------------
static __device__ inline v8f wmma16(v16h a, v16h b, v8f c) {
  // (neg_a, A, neg_b, B, c_mod, C, reuse_a, reuse_b)
  return __builtin_amdgcn_wmma_f32_16x16x32_f16(false, a, false, b, (short)0, c,
                                                false, false);
}

// A-matrix 16x32 f16 fragment from row-major [16 x >=32] with leading dim ld.
// Lane layout (ISA 7.12.2): lane m=lane&15, half=lane>>4;
//   elems 0..7  -> k =      8*half + e ; elems 8..15 -> k = 16 + 8*half + e-8
static __device__ inline v16h load_a_frag(const _Float16* __restrict__ p, int ld,
                                          int lane) {
  const int m = lane & 15, half = lane >> 4;
  const _Float16* r = p + (size_t)m * ld + 8 * half;
  v8h lo = *(const v8h*)(r);        // k = 8*half .. +7
  v8h hi = *(const v8h*)(r + 16);   // k = 16+8*half .. +7
  return __builtin_shufflevector(lo, hi, 0, 1, 2, 3, 4, 5, 6, 7, 8, 9, 10, 11,
                                 12, 13, 14, 15);
}

// B-matrix 32x16 f16 fragment from K-major storage [32 x 16], leading dim ld.
// Lane n = lane&15, half = lane>>4; elems 0..7 -> k = 8*half + e,
// elems 8..15 -> k = 16 + 8*half + e-8 (same k order as A).
static __device__ inline v16h load_b_frag(const _Float16* __restrict__ p, int ld,
                                          int lane) {
  const _Float16* r = p + (size_t)(8 * (lane >> 4)) * ld + (lane & 15);
  v16h v;
#pragma unroll
  for (int e = 0; e < 8; ++e) {
    v[e] = r[(size_t)e * ld];
    v[e + 8] = r[(size_t)(16 + e) * ld];
  }
  return v;
}

// ---------------------------------------------------------------------------
// f32 -> f16 convert (vectorized) / LDS-tiled weight transpose
// ---------------------------------------------------------------------------
__global__ void cvt_f16_v4(const float* __restrict__ in,
                           _Float16* __restrict__ out) {
  const int i = blockIdx.x * blockDim.x + threadIdx.x;  // one float4 each
  const float4 x = ((const float4*)in)[i];
  v4h y = {(_Float16)x.x, (_Float16)x.y, (_Float16)x.z, (_Float16)x.w};
  *(v4h*)(out + 4 * (size_t)i) = y;
}

// Wt[k][n] = (f16) W[n][k]; coalesced read and write via 32x32 LDS tile.
__global__ __launch_bounds__(256) void transpose_cvt_f16(
    const float* __restrict__ W, _Float16* __restrict__ Wt) {
  __shared__ float tile[32][33];
  const int tx = threadIdx.x, ty = threadIdx.y;      // block (32, 8)
  const int bx = blockIdx.x * 32, by = blockIdx.y * 32;
#pragma unroll
  for (int j = 0; j < 32; j += 8)
    tile[ty + j][tx] = W[(size_t)(by + ty + j) * D_MODEL + bx + tx];
  __syncthreads();
#pragma unroll
  for (int j = 0; j < 32; j += 8)
    Wt[(size_t)(bx + ty + j) * D_MODEL + by + tx] = (_Float16)tile[tx][ty + j];
}

// ---------------------------------------------------------------------------
// GEMM: Y[M,N] = A[M,K] * W^T  with Bt = W^T stored K-major f16.
// MODE 0: f16 out * outScale, head-split [B,H,S,Dk]         (K projection)
// MODE 1: f16 out * outScale, head-split transp. [B,H,Dk,S] (Q, V projections)
// MODE 2: f32 out, plain row-major [M,N]                     (W_o -> d_out)
// Block = 4 waves (2x2), wave tile 32x64, block tile 64x128.
// ---------------------------------------------------------------------------
template <int MODE>
__global__ __launch_bounds__(128) void gemm_wmma(
    const _Float16* __restrict__ A, const _Float16* __restrict__ Bt,
    void* __restrict__ Out, int M, int N, int K, float outScale) {
  const int lane = threadIdx.x & 31;
  const int wave = threadIdx.x >> 5;
  const int m0 = blockIdx.x * 64 + (wave & 1) * 32;
  const int n0 = blockIdx.y * 128 + (wave >> 1) * 64;

  v8f acc[2][4] = {};
  for (int k0 = 0; k0 < K; k0 += 32) {
    v16h a0 = load_a_frag(A + (size_t)m0 * K + k0, K, lane);
    v16h a1 = load_a_frag(A + (size_t)(m0 + 16) * K + k0, K, lane);
#pragma unroll
    for (int j = 0; j < 4; ++j) {
      v16h b = load_b_frag(Bt + (size_t)k0 * N + n0 + 16 * j, N, lane);
      acc[0][j] = wmma16(a0, b, acc[0][j]);
      acc[1][j] = wmma16(a1, b, acc[1][j]);
    }
  }

  // C/D layout: lane 0-15 -> N=lane, M=r ; lane 16-31 -> N=lane-16, M=r+8
  const int half = lane >> 4, nl = lane & 15;
#pragma unroll
  for (int i = 0; i < 2; ++i) {
#pragma unroll
    for (int j = 0; j < 4; ++j) {
#pragma unroll
      for (int r = 0; r < 8; ++r) {
        const int g = m0 + i * 16 + r + 8 * half;  // row = b*SEQ + s
        const int col = n0 + 16 * j + nl;          // 0..D_MODEL-1
        const float v = acc[i][j][r];
        if (MODE == 2) {
          ((float*)Out)[(size_t)g * N + col] = v;
        } else {
          const int b = g >> 11, s = g & (SEQ - 1);
          const int h = col >> 6, d = col & (D_K - 1);
          _Float16* o = (_Float16*)Out;
          if (MODE == 0)
            o[((size_t)(b * NUM_HEADS + h) * SEQ + s) * D_K + d] =
                (_Float16)(v * outScale);
          else
            o[((size_t)(b * NUM_HEADS + h) * D_K + d) * SEQ + s] =
                (_Float16)(v * outScale);
        }
      }
    }
  }
}

// ---------------------------------------------------------------------------
// Flash attention in TRANSPOSED score space: S^T = K * Q^T (keys in rows,
// queries in lanes), so the softmax over keys is a pure per-lane reduction
// (no cross-lane ops at all). O^T = V^T * P^T accumulates d-major, giving a
// vectorized 16B/frag epilogue store. Q is pre-scaled by 1/sqrt(Dk).
//   Qt:[B,H,Dk,S] (scaled), Kh:[B,H,S,Dk], Vt:[B,H,Dk,S], out f16 [B,S,1024].
// One wave owns 16 queries; block = 4 waves = 64 queries.
// ---------------------------------------------------------------------------
__global__ __launch_bounds__(128) void attention_wmma(
    const _Float16* __restrict__ Qt, const _Float16* __restrict__ Kh,
    const _Float16* __restrict__ Vt, _Float16* __restrict__ attOut) {
  __shared__ _Float16 pbuf[4][32 * 16];  // per-wave P^T tile: 32 keys x 16 q

  const int lane = threadIdx.x & 31;
  const int wave = threadIdx.x >> 5;
  const int bh = blockIdx.y;  // b*NUM_HEADS + h
  const int b = bh >> 4, h = bh & (NUM_HEADS - 1);
  const int q0 = (blockIdx.x * 4 + wave) * 16;
  const int half = lane >> 4, nl = lane & 15;

  const _Float16* Qbase = Qt + (size_t)bh * D_K * SEQ + q0;
  const _Float16* Kbase = Kh + (size_t)bh * SEQ * D_K;
  const _Float16* Vbase = Vt + (size_t)bh * D_K * SEQ;

  // Q^T B-fragments (d-chunks 0 and 32): loaded once, reused for all keys.
  const v16h qb0 = load_b_frag(Qbase, SEQ, lane);
  const v16h qb1 = load_b_frag(Qbase + (size_t)32 * SEQ, SEQ, lane);

  float rm = -__builtin_inff(), rl = 0.f;  // per-lane (= per-query) stats
  v8f o[4] = {};                           // O^T: d rows x query lanes
  _Float16* pb = pbuf[wave];

  for (int kb = 0; kb < SEQ; kb += 32) {
    // ---- S^T tiles: st0 = keys [kb,kb+16), st1 = keys [kb+16,kb+32)
    v8f st0 = {}, st1 = {};
    {
      v16h ak = load_a_frag(Kbase + (size_t)kb * D_K, D_K, lane);
      st0 = wmma16(ak, qb0, st0);
      ak = load_a_frag(Kbase + (size_t)kb * D_K + 32, D_K, lane);
      st0 = wmma16(ak, qb1, st0);
      ak = load_a_frag(Kbase + (size_t)(kb + 16) * D_K, D_K, lane);
      st1 = wmma16(ak, qb0, st1);
      ak = load_a_frag(Kbase + (size_t)(kb + 16) * D_K + 32, D_K, lane);
      st1 = wmma16(ak, qb1, st1);
    }
    // ---- per-lane online softmax over the 16 key scores in this lane
    float t = fmaxf(st0[0], st1[0]);
#pragma unroll
    for (int r = 1; r < 8; ++r) t = fmaxf(t, fmaxf(st0[r], st1[r]));
    const float mn = fmaxf(rm, t);
    const float alpha = __expf(rm - mn);
    rm = mn;
    float sum = 0.f;
#pragma unroll
    for (int r = 0; r < 8; ++r) {
      st0[r] = __expf(st0[r] - mn);
      st1[r] = __expf(st1[r] - mn);
      sum += st0[r] + st1[r];
    }
    rl = rl * alpha + sum;
#pragma unroll
    for (int f = 0; f < 4; ++f)
#pragma unroll
      for (int r = 0; r < 8; ++r) o[f][r] *= alpha;

    // ---- P^T (C-layout) -> LDS [key row][query col] for B-operand reload
#pragma unroll
    for (int r = 0; r < 8; ++r) {
      pb[(r + 8 * half) * 16 + nl] = (_Float16)st0[r];
      pb[(16 + r + 8 * half) * 16 + nl] = (_Float16)st1[r];
    }
    __syncthreads();
    const v16h pB = load_b_frag(pb, 16, lane);  // lane = query, elems = keys
    // ---- O^T += V^T * P^T
#pragma unroll
    for (int f = 0; f < 4; ++f) {
      v16h av = load_a_frag(Vbase + (size_t)(f * 16) * SEQ + kb, SEQ, lane);
      o[f] = wmma16(av, pB, o[f]);
    }
    __syncthreads();
  }

  // ---- normalize; merge heads: 4 x 16B vector stores per lane
  const float inv = __builtin_amdgcn_rcpf(rl);
  const int s = q0 + nl;  // this lane's query row
  _Float16* orow =
      attOut + (size_t)(b * SEQ + s) * D_MODEL + h * D_K + 8 * half;
#pragma unroll
  for (int f = 0; f < 4; ++f) {
    v8h y;
#pragma unroll
    for (int r = 0; r < 8; ++r) y[r] = (_Float16)(o[f][r] * inv);
    *(v8h*)(orow + f * 16) = y;
  }
}

// ---------------------------------------------------------------------------
extern "C" void kernel_launch(void* const* d_in, const int* in_sizes, int n_in,
                              void* d_out, int out_size, void* d_ws,
                              size_t ws_size, hipStream_t stream) {
  (void)in_sizes; (void)n_in; (void)out_size; (void)ws_size;

  const float* q  = (const float*)d_in[0];
  const float* k  = (const float*)d_in[1];
  const float* v  = (const float*)d_in[2];
  const float* Wq = (const float*)d_in[3];
  const float* Wk = (const float*)d_in[4];
  const float* Wv = (const float*)d_in[5];
  const float* Wo = (const float*)d_in[6];

  const size_t NX = (size_t)M_ROWS * D_MODEL;   // 4M elems
  const size_t NW = (size_t)D_MODEL * D_MODEL;  // 1M elems

  _Float16* ws  = (_Float16*)d_ws;              // 56 MB total f16 staging
  _Float16* Xq  = ws;                           // reused as attOut later
  _Float16* Xk  = Xq + NX;
  _Float16* Xv  = Xk + NX;
  _Float16* Wtq = Xv + NX;
  _Float16* Wtk = Wtq + NW;
  _Float16* Wtv = Wtk + NW;
  _Float16* Wto = Wtv + NW;
  _Float16* QtB = Wto + NW;   // [B,H,Dk,S] (pre-scaled by 1/sqrt(Dk))
  _Float16* KhB = QtB + NX;   // [B,H,S,Dk]
  _Float16* VtB = KhB + NX;   // [B,H,Dk,S]
  _Float16* attO = Xq;        // alias: Xq dead after Q projection

  // Stage f16 copies of activations and K-major f16 weights
  cvt_f16_v4<<<(int)(NX / 1024), 256, 0, stream>>>(q, Xq);
  cvt_f16_v4<<<(int)(NX / 1024), 256, 0, stream>>>(k, Xk);
  cvt_f16_v4<<<(int)(NX / 1024), 256, 0, stream>>>(v, Xv);
  dim3 tb(32, 8), tg(D_MODEL / 32, D_MODEL / 32);
  transpose_cvt_f16<<<tg, tb, 0, stream>>>(Wq, Wtq);
  transpose_cvt_f16<<<tg, tb, 0, stream>>>(Wk, Wtk);
  transpose_cvt_f16<<<tg, tb, 0, stream>>>(Wv, Wtv);
  transpose_cvt_f16<<<tg, tb, 0, stream>>>(Wo, Wto);

  // Projections (Q carries the 1/sqrt(d_k) softmax scale; Q,V are d-major)
  const float qscale = 0.125f;  // 1/sqrt(64)
  dim3 gg(M_ROWS / 64, D_MODEL / 128);  // (64, 8)
  gemm_wmma<1><<<gg, 128, 0, stream>>>(Xq, Wtq, QtB, M_ROWS, D_MODEL, D_MODEL,
                                       qscale);
  gemm_wmma<0><<<gg, 128, 0, stream>>>(Xk, Wtk, KhB, M_ROWS, D_MODEL, D_MODEL,
                                       1.0f);
  gemm_wmma<1><<<gg, 128, 0, stream>>>(Xv, Wtv, VtB, M_ROWS, D_MODEL, D_MODEL,
                                       1.0f);

  // Flash attention (transposed score space)
  dim3 ga(SEQ / 64, BATCH * NUM_HEADS);  // (32, 32)
  attention_wmma<<<ga, 128, 0, stream>>>(QtB, KhB, VtB, attO);

  // Output projection -> f32 d_out [B,S,D_MODEL]
  gemm_wmma<2><<<gg, 128, 0, stream>>>(attO, Wto, d_out, M_ROWS, D_MODEL,
                                       D_MODEL, 1.0f);
}
